// CausalSelfAttention_35476429865020
// MI455X (gfx1250) — matrix operand order repaired and checked by reading, not compiled, not run on hardware
//
#include <hip/hip_runtime.h>
#include <hip/hip_bf16.h>

typedef __bf16 bf16_t;
typedef __attribute__((ext_vector_type(16))) __bf16 v16bf;
typedef __attribute__((ext_vector_type(8)))  float  v8f;
typedef __attribute__((ext_vector_type(4)))  int    v4i;
typedef __attribute__((ext_vector_type(8)))  int    v8i;

#define DIM_    2048
#define SEQ_    2048
#define BATCH_  2
#define NHEADS_ 16
#define NKV_    4
#define HDIM_   128
#define KVDIM_  512
#define MTOT_   (BATCH_*SEQ_)
#define RMS_EPS_ 1.1920928955078125e-07f

// ---------------------------------------------------------------- CDNA5 async global->LDS copy
__device__ __forceinline__ void async_copy_b128(unsigned lds_off, const bf16_t* gptr) {
    asm volatile("global_load_async_to_lds_b128 %0, %1, off"
                 :
                 : "v"(lds_off), "v"(gptr)
                 : "memory");
}
__device__ __forceinline__ void wait_async() {
    asm volatile("s_wait_asynccnt 0" ::: "memory");
}
// LDS aperture keeps the wave-relative LDS offset in addr[31:0] (ISA 10.2 aperture mapping).
__device__ __forceinline__ unsigned lds_addr(const void* p) {
    return (unsigned)(size_t)p;
}

// ---------------------------------------------------------------- CDNA5 Tensor Data Mover 2D tile load
// D# per ISA 08_async_tensor 8.3/8.4: group0 = {count/type, lds_addr, global_addr},
// group1 = {data_size=2B, tensor dims, tile dims, dim0 stride}. 2-group form (<=2D tensor).
// tile_w elems per row -> packed contiguously in LDS (row y at lds + y*tile_w*2).
__device__ __forceinline__ void tdm_load_2d(unsigned lds, const void* gaddr,
                                            unsigned tile_w, unsigned tile_h,
                                            unsigned stride_elems)
{
    const unsigned long long ga = (unsigned long long)gaddr;
    v4i g0;
    g0[0] = 1;                                            // count=1 (valid user descriptor)
    g0[1] = (int)lds;                                     // lds_addr [63:32]
    g0[2] = (int)(ga & 0xFFFFFFFFu);                      // global_addr [95:64]
    g0[3] = (int)((ga >> 32) & 0x01FFFFFFu) | (2 << 30);  // global_addr[56:32] | type=2
    v8i g1;
    g1[0] = 1 << 16;                                      // data_size=1 (2 bytes)
    g1[1] = (int)((tile_w & 0xFFFFu) << 16);              // tensor_dim0[15:0] @ bit48
    g1[2] = (int)((tile_w >> 16) | ((tile_h & 0xFFFFu) << 16));   // tensor_dim0[31:16], tensor_dim1[15:0]
    g1[3] = (int)((tile_h >> 16) | ((tile_w & 0xFFFFu) << 16));   // tensor_dim1[31:16], tile_dim0
    g1[4] = (int)(tile_h & 0xFFFFu);                      // tile_dim1 (tile_dim2=0)
    g1[5] = (int)stride_elems;                            // tensor_dim0_stride[31:0]
    g1[6] = 0;                                            // stride0 hi, tensor_dim1_stride lo
    g1[7] = 0;
    asm volatile("tensor_load_to_lds %0, %1"
                 :
                 : "s"(g0), "s"(g1)
                 : "memory");
}

// ---------------------------------------------------------------- cast f32 -> bf16
__global__ void cast_f32_to_bf16(const float* __restrict__ in, bf16_t* __restrict__ out, int n) {
    int i = blockIdx.x * blockDim.x + threadIdx.x;
    if (i < n) out[i] = (bf16_t)in[i];
}

// ---------------------------------------------------------------- GEMM  C[M,N] = A[M,K] * W[N,K]^T
// 128x128 block tile, 256 threads = 8 waves (wave32), wave tile 64x32 = 4x2 WMMA frags.
// Double-buffered LDS staging via async global->LDS copies (copy of tile t+1 hides under WMMAs of tile t).
__launch_bounds__(256)
__global__ void gemm_bf16_nt(const bf16_t* __restrict__ A,
                             const bf16_t* __restrict__ W,
                             float* __restrict__ C,
                             int M, int N, int K)
{
    __shared__ bf16_t lds_a[2][128][32];
    __shared__ bf16_t lds_b[2][128][32];

    const int tid  = threadIdx.x;
    const int wave = tid >> 5;
    const int lane = tid & 31;
    const int l16  = lane & 15;
    const int hi   = lane >> 4;

    const int wm = wave >> 2;      // 0..1 -> 64 M rows
    const int wn = wave & 3;       // 0..3 -> 32 N cols

    const int m0 = blockIdx.y * 128;
    const int n0 = blockIdx.x * 128;

    v8f acc[4][2];
    #pragma unroll
    for (int i = 0; i < 4; i++)
        #pragma unroll
        for (int j = 0; j < 2; j++)
            #pragma unroll
            for (int r = 0; r < 8; r++) acc[i][j][r] = 0.0f;

    const int ldRow  = tid >> 1;         // 0..127
    const int ldHalf = (tid & 1) * 16;   // 0 or 16

    auto stage = [&](int buf, int k0) {
        const bf16_t* ap = A + (size_t)(m0 + ldRow) * K + k0 + ldHalf;   // 32 contiguous bytes
        const bf16_t* wp = W + (size_t)(n0 + ldRow) * K + k0 + ldHalf;
        const unsigned la = lds_addr(&lds_a[buf][ldRow][ldHalf]);
        const unsigned lb = lds_addr(&lds_b[buf][ldRow][ldHalf]);
        async_copy_b128(la,      ap);
        async_copy_b128(la + 16, ap + 8);
        async_copy_b128(lb,      wp);
        async_copy_b128(lb + 16, wp + 8);
    };

    stage(0, 0);
    int buf = 0;
    for (int k0 = 0; k0 < K; k0 += 32) {
        wait_async();
        __syncthreads();
        if (k0 + 32 < K) stage(buf ^ 1, k0 + 32);   // prefetch next tile under the WMMAs

        // A fragment: 16x32, lane l (row = l%16) holds K = {base..+7, 16+base..+7}, base = 8*(l>=16)
        v16bf afrag[4];
        #pragma unroll
        for (int mi = 0; mi < 4; mi++) {
            const int row = wm * 64 + mi * 16 + l16;
            const int kb  = hi * 8;
            #pragma unroll
            for (int e = 0; e < 8; e++) {
                afrag[mi][e]     = lds_a[buf][row][kb + e];
                afrag[mi][8 + e] = lds_a[buf][row][16 + kb + e];
            }
        }
        // B fragment: 32x16, lane l (col n = l%16) holds K = {base..+7, 16+base..+7}, base = 8*(l>=16)
        v16bf bfrag[2];
        #pragma unroll
        for (int ni = 0; ni < 2; ni++) {
            const int col = wn * 32 + ni * 16 + l16;
            const int k2  = hi * 8;
            #pragma unroll
            for (int e = 0; e < 8; e++) {
                bfrag[ni][e]     = lds_b[buf][col][k2 + e];
                bfrag[ni][8 + e] = lds_b[buf][col][16 + k2 + e];
            }
        }
        #pragma unroll
        for (int mi = 0; mi < 4; mi++)
            #pragma unroll
            for (int ni = 0; ni < 2; ni++)
                acc[mi][ni] = __builtin_amdgcn_wmma_f32_16x16x32_bf16(
                    false, afrag[mi], false, bfrag[ni],
                    (short)0, acc[mi][ni], false, false);
        buf ^= 1;
    }

    // C/D layout: VGPR r, lane l -> row = r + 8*(l>=16), col = l%16
    #pragma unroll
    for (int mi = 0; mi < 4; mi++)
        #pragma unroll
        for (int ni = 0; ni < 2; ni++) {
            const int col   = n0 + wn * 32 + ni * 16 + l16;
            const int rbase = m0 + wm * 64 + mi * 16 + hi * 8;
            #pragma unroll
            for (int r = 0; r < 8; r++)
                C[(size_t)(rbase + r) * N + col] = acc[mi][ni][r];
        }
}

// ---------------------------------------------------------------- RMSNorm + RoPE + gain, f32 -> bf16
__launch_bounds__(128)
__global__ void rms_rope_cast(const float* __restrict__ src,
                              bf16_t* __restrict__ dst,
                              const float* __restrict__ gain,
                              int nheads, int rowdim, int use_gain)
{
    __shared__ float xs[HDIM_];
    __shared__ float red[HDIM_];
    const int d   = threadIdx.x;
    const int h   = blockIdx.x % nheads;
    const int tok = blockIdx.x / nheads;

    const float v = src[(size_t)tok * rowdim + h * HDIM_ + d];
    xs[d]  = v;
    red[d] = v * v;
    __syncthreads();
    for (int s = 64; s > 0; s >>= 1) {
        if (d < s) red[d] += red[d + s];
        __syncthreads();
    }
    const float rr  = rsqrtf(red[0] * (1.0f / HDIM_) + RMS_EPS_);
    const float pos = (float)(tok % SEQ_);

    float out;
    if (d < 64) {
        const float ang = pos * __powf(10000.0f, -(float)d * (1.0f / 64.0f));
        out = (xs[d] * rr) * __cosf(ang) + (xs[d + 64] * rr) * __sinf(ang);
    } else {
        const int j = d - 64;
        const float ang = pos * __powf(10000.0f, -(float)j * (1.0f / 64.0f));
        out = -(xs[j] * rr) * __sinf(ang) + (xs[d] * rr) * __cosf(ang);
    }
    if (use_gain) out *= gain[h];
    dst[(size_t)tok * rowdim + h * HDIM_ + d] = (bf16_t)out;
}

// ---------------------------------------------------------------- V: cast f32 -> bf16 + transpose to [b][kvh][d][s]
__launch_bounds__(256)
__global__ void v_cast_transpose(const float* __restrict__ vf, bf16_t* __restrict__ vbt)
{
    __shared__ bf16_t tile[64][136];   // 64 s x 128 d, padded
    const int s0  = blockIdx.x * 64;
    const int kvh = blockIdx.y;
    const int b   = blockIdx.z;
    const int t   = threadIdx.x;
    {
        const int sl = t >> 2;          // 0..63
        const int d0 = (t & 3) * 32;    // 0,32,64,96
        const float* src = vf + (size_t)(b * SEQ_ + s0 + sl) * KVDIM_ + kvh * HDIM_ + d0;
        #pragma unroll
        for (int e = 0; e < 32; e++)
            tile[sl][d0 + e] = (bf16_t)src[e];
    }
    __syncthreads();
    {
        const int d  = t >> 1;          // 0..127
        const int sh = (t & 1) * 32;    // 0,32
        bf16_t* dst = vbt + (size_t)((b * NKV_ + kvh) * HDIM_ + d) * SEQ_ + s0 + sh;
        #pragma unroll
        for (int e = 0; e < 32; e++)
            dst[e] = tile[sh + e][d];
    }
}

// ---------------------------------------------------------------- flash attention (causal, GQA)
// grid (S/64, NHEADS, BATCH), 128 threads = 4 waves; each wave owns 16 query rows.
// K and V^T tiles staged by the Tensor Data Mover (double-buffered; DMA of tile t+1
// runs under the QK^T/softmax/PV math of tile t, tracked by TENSORcnt).
__launch_bounds__(128)
__global__ void flash_attn_wmma(const bf16_t* __restrict__ qb,
                                const bf16_t* __restrict__ kb,
                                const bf16_t* __restrict__ vbt,
                                bf16_t* __restrict__ yb)
{
    __shared__ bf16_t kt[2][32][HDIM_];    // [key][d]
    __shared__ bf16_t vtT[2][HDIM_][32];   // [d][key]  (pre-transposed in global)
    __shared__ bf16_t pt[4][16][32];

    const int tid  = threadIdx.x;
    const int wave = tid >> 5;
    const int lane = tid & 31;
    const int l16  = lane & 15;
    const int hi   = lane >> 4;

    const int b   = blockIdx.z;
    const int h   = blockIdx.y;
    const int kvh = h / (NHEADS_ / NKV_);
    const int q0  = blockIdx.x * 64 + wave * 16;   // wave's first query row

    // Q fragments: 16 rows x 128 dims = 4 K-steps of 32 (register resident)
    v16bf qf[4];
    {
        const int row = q0 + l16;
        const bf16_t* qp = qb + (size_t)(b * SEQ_ + row) * DIM_ + h * HDIM_;
        #pragma unroll
        for (int ds = 0; ds < 4; ds++) {
            const int kb8 = hi * 16;
            #pragma unroll
            for (int e = 0; e < 16; e++)
                qf[ds][e] = qp[ds * 32 + kb8 + e];
        }
    }

    v8f o[8];
    float rmax[8], rsum[8];
    #pragma unroll
    for (int i = 0; i < 8; i++) {
        rmax[i] = -INFINITY;
        rsum[i] = 0.0f;
        #pragma unroll
        for (int r = 0; r < 8; r++) o[i][r] = 0.0f;
    }

    const int   kend  = blockIdx.x * 64 + 64;   // block-uniform causal bound
    const float scale = 0.08838834764831845f;   // 1/sqrt(128)

    // TDM staging: one wave issues two 2D descriptors per tile (EXEC ignored by TDM).
    auto stage = [&](int buf, int k0) {
        if (wave == 0) {
            tdm_load_2d(lds_addr(&kt[buf][0][0]),
                        kb + (size_t)(b * SEQ_ + k0) * KVDIM_ + kvh * HDIM_,
                        /*tile_w=*/HDIM_, /*tile_h=*/32, /*stride=*/KVDIM_);
            tdm_load_2d(lds_addr(&vtT[buf][0][0]),
                        vbt + (size_t)(b * NKV_ + kvh) * HDIM_ * SEQ_ + k0,
                        /*tile_w=*/32, /*tile_h=*/HDIM_, /*stride=*/SEQ_);
        }
    };

    stage(0, 0);
    int buf = 0;
    for (int k0 = 0; k0 < kend; k0 += 32) {
        __builtin_amdgcn_s_wait_tensorcnt((short)0);
        __syncthreads();
        if (k0 + 32 < kend) stage(buf ^ 1, k0 + 32);   // DMA next tile under this tile's math

        const bool active = (k0 <= q0 + 15);    // wave-uniform -> EXEC all-ones inside
        if (active) {
            v8f sc[2];
            #pragma unroll
            for (int nt = 0; nt < 2; nt++)
                #pragma unroll
                for (int r = 0; r < 8; r++) sc[nt][r] = 0.0f;

            #pragma unroll
            for (int ds = 0; ds < 4; ds++)
                #pragma unroll
                for (int nt = 0; nt < 2; nt++) {
                    v16bf bfr;                      // B = K^T: col n = key, contraction = d
                    const int key = nt * 16 + l16;
                    const int k2  = hi * 16;
                    #pragma unroll
                    for (int e = 0; e < 16; e++)
                        bfr[e] = kt[buf][key][ds * 32 + k2 + e];
                    sc[nt] = __builtin_amdgcn_wmma_f32_16x16x32_bf16(
                        false, qf[ds], false, bfr, (short)0, sc[nt], false, false);
                }

            // online softmax, per C-layout row (row = hi*8 + rI, col = l%16)
            #pragma unroll
            for (int rI = 0; rI < 8; rI++) {
                const int rowg = q0 + hi * 8 + rI;
                float s0 = sc[0][rI] * scale; if (k0 + l16 > rowg)      s0 = -INFINITY;
                float s1 = sc[1][rI] * scale; if (k0 + 16 + l16 > rowg) s1 = -INFINITY;
                float m = fmaxf(s0, s1);
                #pragma unroll
                for (int off = 8; off >= 1; off >>= 1)
                    m = fmaxf(m, __shfl_xor(m, off, 32));
                const float mnew = fmaxf(rmax[rI], m);
                const float corr = __expf(rmax[rI] - mnew);
                const float p0 = __expf(s0 - mnew);
                const float p1 = __expf(s1 - mnew);
                float ps = p0 + p1;
                #pragma unroll
                for (int off = 8; off >= 1; off >>= 1)
                    ps += __shfl_xor(ps, off, 32);
                rsum[rI] = rsum[rI] * corr + ps;
                rmax[rI] = mnew;
                #pragma unroll
                for (int ni = 0; ni < 8; ni++) o[ni][rI] *= corr;
                pt[wave][hi * 8 + rI][l16]      = (bf16_t)p0;
                pt[wave][hi * 8 + rI][16 + l16] = (bf16_t)p1;
            }
        }
        __syncthreads();
        if (active) {
            // P (16x32 keys) as A fragment
            v16bf pf;
            const int kb8 = hi * 16;
            #pragma unroll
            for (int e = 0; e < 16; e++)
                pf[e] = pt[wave][l16][kb8 + e];
            // V (32 keys x 128 dims) as 8 B fragments; contiguous reads from vtT
            #pragma unroll
            for (int ni = 0; ni < 8; ni++) {
                v16bf vf;
                const int n  = ni * 16 + l16;
                const int k2 = hi * 16;
                #pragma unroll
                for (int e = 0; e < 16; e++)
                    vf[e] = vtT[buf][n][k2 + e];
                o[ni] = __builtin_amdgcn_wmma_f32_16x16x32_bf16(
                    false, pf, false, vf, (short)0, o[ni], false, false);
            }
        }
        buf ^= 1;
    }

    #pragma unroll
    for (int ni = 0; ni < 8; ni++)
        #pragma unroll
        for (int rI = 0; rI < 8; rI++) {
            const int rowg = q0 + hi * 8 + rI;
            const float val = o[ni][rI] / rsum[rI];
            yb[(size_t)(b * SEQ_ + rowg) * DIM_ + h * HDIM_ + ni * 16 + l16] = (bf16_t)val;
        }
}

// ---------------------------------------------------------------- launch
extern "C" void kernel_launch(void* const* d_in, const int* in_sizes, int n_in,
                              void* d_out, int out_size, void* d_ws, size_t ws_size,
                              hipStream_t stream)
{
    const float* x  = (const float*)d_in[0];
    const float* Wq = (const float*)d_in[1];
    const float* Wk = (const float*)d_in[2];
    const float* Wv = (const float*)d_in[3];
    const float* Wp = (const float*)d_in[4];
    const float* qg = (const float*)d_in[5];
    float* out = (float*)d_out;

    char* ws = (char*)d_ws;
    size_t off = 0;
    auto alloc = [&](size_t bytes) -> char* {
        char* p = ws + off;
        off += (bytes + 255) & ~(size_t)255;
        return p;
    };

    bf16_t* xb  = (bf16_t*)alloc((size_t)MTOT_ * DIM_ * 2);
    bf16_t* wqb = (bf16_t*)alloc((size_t)DIM_ * DIM_ * 2);
    bf16_t* wkb = (bf16_t*)alloc((size_t)KVDIM_ * DIM_ * 2);
    bf16_t* wvb = (bf16_t*)alloc((size_t)KVDIM_ * DIM_ * 2);
    bf16_t* wpb = (bf16_t*)alloc((size_t)DIM_ * DIM_ * 2);
    float*  qf  = (float*) alloc((size_t)MTOT_ * DIM_ * 4);
    float*  kf  = (float*) alloc((size_t)MTOT_ * KVDIM_ * 4);
    float*  vf  = (float*) alloc((size_t)MTOT_ * KVDIM_ * 4);
    bf16_t* qbn = (bf16_t*)alloc((size_t)MTOT_ * DIM_ * 2);
    bf16_t* kbn = (bf16_t*)alloc((size_t)MTOT_ * KVDIM_ * 2);
    bf16_t* vbt = (bf16_t*)alloc((size_t)MTOT_ * KVDIM_ * 2);   // [b][kvh][d][s]
    bf16_t* yb  = (bf16_t*)alloc((size_t)MTOT_ * DIM_ * 2);

    auto castN = [&](const float* src, bf16_t* dst, int n) {
        cast_f32_to_bf16<<<(n + 255) / 256, 256, 0, stream>>>(src, dst, n);
    };
    castN(x,  xb,  MTOT_ * DIM_);
    castN(Wq, wqb, DIM_ * DIM_);
    castN(Wk, wkb, KVDIM_ * DIM_);
    castN(Wv, wvb, KVDIM_ * DIM_);
    castN(Wp, wpb, DIM_ * DIM_);

    gemm_bf16_nt<<<dim3(DIM_ / 128,   MTOT_ / 128), 256, 0, stream>>>(xb, wqb, qf, MTOT_, DIM_,   DIM_);
    gemm_bf16_nt<<<dim3(KVDIM_ / 128, MTOT_ / 128), 256, 0, stream>>>(xb, wkb, kf, MTOT_, KVDIM_, DIM_);
    gemm_bf16_nt<<<dim3(KVDIM_ / 128, MTOT_ / 128), 256, 0, stream>>>(xb, wvb, vf, MTOT_, KVDIM_, DIM_);

    rms_rope_cast<<<MTOT_ * NHEADS_, 128, 0, stream>>>(qf, qbn, qg, NHEADS_, DIM_,   1);
    rms_rope_cast<<<MTOT_ * NKV_,    128, 0, stream>>>(kf, kbn, qg, NKV_,    KVDIM_, 0);
    v_cast_transpose<<<dim3(SEQ_ / 64, NKV_, BATCH_), 256, 0, stream>>>(vf, vbt);

    flash_attn_wmma<<<dim3(SEQ_ / 64, NHEADS_, BATCH_), 128, 0, stream>>>(qbn, kbn, vbt, yb);

    gemm_bf16_nt<<<dim3(DIM_ / 128, MTOT_ / 128), 256, 0, stream>>>(yb, wpb, out, MTOT_, DIM_, DIM_);

    (void)in_sizes; (void)n_in; (void)out_size; (void)ws_size;
}
